// FastSparseLinearSelfAttention_52226802319938
// MI455X (gfx1250) — matrix operand order repaired and checked by reading, not compiled, not run on hardware
//
#include <hip/hip_runtime.h>

// ---------------------------------------------------------------------------
// MI455X (gfx1250) fused linear self-attention.
// Roofline: 19.4 GFLOP, 268 MB min HBM traffic -> 11.5us floor @23.3TB/s.
// bf16 WMMA (v_wmma_f32_16x16x32_bf16, fp32 acc) for every GEMM.
// This revision: zero-fragment ADDRESS selection instead of per-dword data
// masking (1 cndmask instead of 8 per fragment), and packed f32->bf16
// conversion via v_cvt_pk_bf16_f32 when the builtin exists.
// ---------------------------------------------------------------------------

typedef __attribute__((ext_vector_type(16))) __bf16   v16bf;
typedef __attribute__((ext_vector_type(8)))  float    v8f;
typedef __attribute__((ext_vector_type(8)))  unsigned v8u;

#define C_DIM   128
#define HW_FULL 65536      // 256*256
#define EPS_F   1e-6f

#if __has_builtin(__builtin_amdgcn_cvt_pk_bf16_f32)
typedef __attribute__((ext_vector_type(2))) __bf16 v2bf;
static __device__ __forceinline__ unsigned packbf(float lo, float hi) {
  v2bf p = __builtin_amdgcn_cvt_pk_bf16_f32(lo, hi);
  return __builtin_bit_cast(unsigned, p);
}
static __device__ __forceinline__ unsigned bfbits(float f) {
  return packbf(f, 0.f) & 0xffffu;
}
#else
static __device__ __forceinline__ unsigned bfbits(float f) {
  unsigned u = __builtin_bit_cast(unsigned, f);
  return (u + 0x7fffu + ((u >> 16) & 1u)) >> 16;   // f32->bf16 RNE
}
static __device__ __forceinline__ unsigned packbf(float lo, float hi) {
  return bfbits(lo) | (bfbits(hi) << 16);
}
#endif

static __device__ __forceinline__ float elu1(float x) {
  return x > 0.f ? x + 1.f : __expf(x);             // elu(x)+1
}
static __device__ __forceinline__ v8f vzero8() {
  v8f z;
#pragma unroll
  for (int i = 0; i < 8; ++i) z[i] = 0.f;
  return z;
}
// 32B fragment load from LDS: two ds_load_b128 (entry is 16B aligned)
static __device__ __forceinline__ v8u ldsv8(const unsigned* p) {
  uint4 a = *reinterpret_cast<const uint4*>(p);
  uint4 b = *reinterpret_cast<const uint4*>(p + 4);
  v8u r;
  r[0] = a.x; r[1] = a.y; r[2] = a.z; r[3] = a.w;
  r[4] = b.x; r[5] = b.y; r[6] = b.z; r[7] = b.w;
  return r;
}
static __device__ __forceinline__ v8f wmma_bf16(v8u a, v8u b, v8f c) {
  return __builtin_amdgcn_wmma_f32_16x16x32_bf16(
      false, __builtin_bit_cast(v16bf, a),
      false, __builtin_bit_cast(v16bf, b),
      (short)0, c, false, false);
}

// ---------------------------------------------------------------------------
// prep: pack weights to bf16-pair u32 tables, zero KV / Ksum accumulators.
//   qwT  [n][kp] : q weights, out-channel major (A-frag source)   (128 x 64)
//   kvwpk[kp][n] : kv weights, k-pair major     (B-frag source)   (64 x 256)
//   pwT  [n][kp] : proj weights, out-channel major (A-frag src)   (128 x 64)
// ---------------------------------------------------------------------------
__global__ __launch_bounds__(256) void prep_kernel(
    const float* __restrict__ q_w, const float* __restrict__ kv_w,
    const float* __restrict__ p_w,
    unsigned* __restrict__ qwT, unsigned* __restrict__ kvwpk,
    unsigned* __restrict__ pwT, float* __restrict__ KVg, float* __restrict__ Ksg)
{
  int i = blockIdx.x * 256 + threadIdx.x;
  if (i < 8192) {
    int n = i >> 6, kp = i & 63;
    qwT[i] = packbf(q_w[n * 128 + 2 * kp], q_w[n * 128 + 2 * kp + 1]);
  } else if (i < 24576) {
    int e = i - 8192, kp = e >> 8, n = e & 255;
    kvwpk[e] = packbf(kv_w[n * 128 + 2 * kp], kv_w[n * 128 + 2 * kp + 1]);
  } else if (i < 32768) {
    int e = i - 24576, n = e >> 6, kp = e & 63;
    pwT[e] = packbf(p_w[n * 128 + 2 * kp], p_w[n * 128 + 2 * kp + 1]);
  } else if (i < 40960) {
    KVg[i - 32768] = 0.f;
  } else if (i < 41472) {
    Ksg[i - 40960] = 0.f;
  }
}

// ---------------------------------------------------------------------------
// stage1: kv GEMM on subsampled pixels (x[:, :, 2::4, 2::4]) + per-head
//         KV = K^T V (WMMA, pixel-K) and Ksum accumulation.
// kv tile is stored TRANSPOSED in LDS: kvT[channel][pixel], stride 24 hw.
// ---------------------------------------------------------------------------
__global__ __launch_bounds__(128) void stage1_kernel(
    const float* __restrict__ x, const unsigned* __restrict__ kvwpk,
    const float* __restrict__ kv_b, float* __restrict__ KVg,
    float* __restrict__ Ksg)
{
  extern __shared__ char smem[];
  unsigned* kvf  = (unsigned*)smem;                        // 98304 B frag-major wts
  float*    kvbl = (float*)(smem + 98304);                 // 1024 B
  unsigned short* tiles = (unsigned short*)(smem + 99328); // 4 x 12288 B kvT
  float*    ksl  = (float*)(smem + 148480);                // 4096 B
  unsigned* zf   = (unsigned*)(smem + 152576);             // 32 B of zeros

  const int lane = threadIdx.x & 31, wave = threadIdx.x >> 5;
  const int half = lane >> 4, l16 = lane & 15;
  const int b = blockIdx.x >> 3, wg = blockIdx.x & 7;

  for (int i = threadIdx.x; i < 64 * 256; i += 128) {
    int kp = i >> 8, n = i & 255;
    int kc = kp >> 4, half_ = (kp >> 3) & 1, j = kp & 7;
    int t = n >> 4, lw = n & 15;
    kvf[((((kc * 2 + half_) * 16 + t) * 16 + lw) * 12) + j] = kvwpk[i];
  }
  for (int i = threadIdx.x; i < 256; i += 128) kvbl[i] = kv_b[i];
  if (threadIdx.x < 8) zf[threadIdx.x] = 0u;
  __syncthreads();

  unsigned short* kvT = tiles + wave * (256 * 24);
  float* ksw = ksl + wave * 256;
#pragma unroll
  for (int t = 0; t < 8; ++t) ksw[t * 32 + lane] = 0.f;

  const float* xb = x + (size_t)b * C_DIM * HW_FULL;
  v8f kvD[8];
#pragma unroll
  for (int h = 0; h < 8; ++h) kvD[h] = vzero8();

  for (int tile = wg * 4 + wave; tile < 256; tile += 32) {
    int sp  = tile * 16 + l16;
    int pix = ((sp >> 6) * 4 + 2) * 256 + ((sp & 63) * 4 + 2);

    // A fragments (16px x 32ch, bf16) straight from global x
    v8u af[4];
#pragma unroll
    for (int kc = 0; kc < 4; ++kc)
#pragma unroll
      for (int j = 0; j < 8; ++j) {
        int ch = kc * 32 + 2 * j + half * 16;
        af[kc][j] = packbf(xb[(size_t)ch * HW_FULL + pix],
                           xb[(size_t)(ch + 1) * HW_FULL + pix]);
      }
    // kv GEMM: 16 N-tiles (8 k-heads, 8 v-heads) x 4 K-chunks
#pragma unroll 2
    for (int t = 0; t < 16; ++t) {
      v8f acc = vzero8();
#pragma unroll
      for (int kc = 0; kc < 4; ++kc) {
        v8u bu = ldsv8(kvf + (((kc * 2 + half) * 16 + t) * 16 + l16) * 12);
        acc = wmma_bf16(af[kc], bu, acc);
      }
      float bias = kvbl[t * 16 + l16];
      float v0, v1, v2, v3, v4, v5, v6, v7;
      if (t < 8) {
        v0 = elu1(acc[0] + bias); v1 = elu1(acc[1] + bias);
        v2 = elu1(acc[2] + bias); v3 = elu1(acc[3] + bias);
        v4 = elu1(acc[4] + bias); v5 = elu1(acc[5] + bias);
        v6 = elu1(acc[6] + bias); v7 = elu1(acc[7] + bias);
        ksw[t * 32 + lane] += (v0 + v1 + v2 + v3) + (v4 + v5 + v6 + v7);
      } else {
        v0 = acc[0] + bias; v1 = acc[1] + bias; v2 = acc[2] + bias;
        v3 = acc[3] + bias; v4 = acc[4] + bias; v5 = acc[5] + bias;
        v6 = acc[6] + bias; v7 = acc[7] + bias;
      }
      uint4 st;
      st.x = packbf(v0, v1); st.y = packbf(v2, v3);
      st.z = packbf(v4, v5); st.w = packbf(v6, v7);
      // transposed store: row = channel t*16+l16, halfwords = pixels half*8..+7
      *reinterpret_cast<uint4*>(kvT + (t * 16 + l16) * 24 + half * 8) = st;
    }
    // KV_h += K^T V   (M=d, N=v, K=pixel zero-padded to 32)
#pragma unroll
    for (int h = 0; h < 8; ++h) {
      v8u au = ldsv8(half ? zf : (const unsigned*)(kvT + (h * 16 + l16) * 24));
      // B K=16..31 lives in lanes16-31 -> those lanes read the zero fragment
      const unsigned* bsrc = half ? zf
                                  : (const unsigned*)(kvT + (128 + h * 16 + l16) * 24);
      kvD[h] = wmma_bf16(au, ldsv8(bsrc), kvD[h]);
    }
  }
  // flush partials (unnormalized KV: the /S and *S of the reference cancel)
  float* KVb = KVg + b * 2048;
#pragma unroll
  for (int h = 0; h < 8; ++h)
#pragma unroll
    for (int r = 0; r < 8; ++r)
      unsafeAtomicAdd(&KVb[(h * 16 + half * 8 + r) * 16 + l16], kvD[h][r]);
  float* Ksb = Ksg + b * 128;
#pragma unroll
  for (int t = 0; t < 8; ++t)
    unsafeAtomicAdd(&Ksb[t * 16 + l16], ksw[t * 32 + lane]);
}

// ---------------------------------------------------------------------------
// stage2: fused q^T GEMM -> feature map (in registers) -> attention WMMAs
//         (vs KV and vs broadcast-Ksum for the denominator) -> proj GEMM.
// q GEMM computed as q^T = W x x^T: D-tile lane holds Q[pixel=l16][d=8h+r],
// which IS the attention A-fragment -> no Q LDS round-trip at all.
// proj as out^T = W x attn^T -> pixel-coalesced global stores.
// ---------------------------------------------------------------------------
__global__ __launch_bounds__(128) void stage2_kernel(
    const float* __restrict__ x, const unsigned* __restrict__ qwT,
    const unsigned* __restrict__ pwT, const float* __restrict__ KVg,
    const float* __restrict__ Ksg, const float* __restrict__ q_b,
    const float* __restrict__ p_b, float* __restrict__ out)
{
  extern __shared__ char smem[];
  unsigned* qfa   = (unsigned*)smem;                // 49152 B A-frag-major q wts
  unsigned* pwf   = (unsigned*)(smem + 49152);      // 49152 B A-frag-major proj wts
  unsigned* kvpk2 = (unsigned*)(smem + 98304);      // 6144 B [h][v][12]
  unsigned* kspk  = (unsigned*)(smem + 104448);     // 256 B  [h][8]
  float*    qbl   = (float*)(smem + 104704);        // 512 B
  float*    pbl   = (float*)(smem + 105216);        // 512 B
  unsigned* zf    = (unsigned*)(smem + 105728);     // 32 B of zeros
  unsigned short* tiles = (unsigned short*)(smem + 105760); // 4 x 4352 B attn tile

  const int lane = threadIdx.x & 31, wave = threadIdx.x >> 5;
  const int half = lane >> 4, l16 = lane & 15;
  const int b = blockIdx.x >> 7, wg = blockIdx.x & 127;

  for (int i = threadIdx.x; i < 8192; i += 128) {
    // A-frag-major rearrange (kp = kc*16 + 8*half + j)
    int n = i >> 6, kp = i & 63;
    int kc = kp >> 4, r = kp & 15;
    int half_ = r >> 3, j = r & 7;
    int t = n >> 4, lw = n & 15;
    int dst = ((((kc * 2 + half_) * 8 + t) * 16 + lw) * 12) + j;
    qfa[dst] = qwT[i];
    pwf[dst] = pwT[i];
  }
  const float* KVb = KVg + b * 2048;
  const float* Ksb = Ksg + b * 128;
  for (int i = threadIdx.x; i < 1024; i += 128) {
    int h = i >> 7, j = (i >> 4) & 7, v = i & 15;
    kvpk2[(h * 16 + v) * 12 + j] = packbf(KVb[(h * 16 + 2 * j) * 16 + v],
                                          KVb[(h * 16 + 2 * j + 1) * 16 + v]);
  }
  if (threadIdx.x < 64) {
    int h = threadIdx.x >> 3, j = threadIdx.x & 7;
    kspk[threadIdx.x] = packbf(Ksb[h * 16 + 2 * j], Ksb[h * 16 + 2 * j + 1]);
  }
  if (threadIdx.x < 128) { qbl[threadIdx.x] = q_b[threadIdx.x]; pbl[threadIdx.x] = p_b[threadIdx.x]; }
  if (threadIdx.x < 8) zf[threadIdx.x] = 0u;
  __syncthreads();

  unsigned short* atile = tiles + wave * (16 * 136);
  const float* xb = x + (size_t)b * C_DIM * HW_FULL;
  float* ob = out + (size_t)b * C_DIM * HW_FULL;

  const int gw = wg * 4 + wave;
  for (int tile = gw; tile < 4096; tile += 512) {
    const int pixbase = tile * 16;
    const int pix = pixbase + l16;

    if (tile + 512 < 4096) {                     // gfx1250 global_prefetch of next tile
#pragma unroll
      for (int kc = 0; kc < 4; ++kc)
        __builtin_prefetch(xb + (size_t)(kc * 32 + half * 16) * HW_FULL + pix + 512 * 16, 0, 0);
    }

    // B fragments x^T (K=in ch, N=pixel): coalesced f32 loads -> bf16 pairs
    v8u bx[4];
#pragma unroll
    for (int kc = 0; kc < 4; ++kc)
#pragma unroll
      for (int j = 0; j < 8; ++j) {
        int ch = kc * 32 + 2 * j + 16 * half;
        bx[kc][j] = packbf(xb[(size_t)ch * HW_FULL + pix],
                           xb[(size_t)(ch + 1) * HW_FULL + pix]);
      }
    // ---- q^T GEMM -> feature map -> attention, all register-resident ----
#pragma unroll 2
    for (int t = 0; t < 8; ++t) {
      v8f acc = vzero8();
#pragma unroll
      for (int kc = 0; kc < 4; ++kc) {
        v8u aw = ldsv8(qfa + (((kc * 2 + half) * 8 + t) * 16 + l16) * 12);
        acc = wmma_bf16(aw, bx[kc], acc);
      }
      // bias along M (rows = head dims): two float4 LDS loads
      float4 b0 = *reinterpret_cast<const float4*>(qbl + t * 16 + half * 8);
      float4 b1 = *reinterpret_cast<const float4*>(qbl + t * 16 + half * 8 + 4);
      float q0 = elu1(acc[0] + b0.x), q1 = elu1(acc[1] + b0.y);
      float q2 = elu1(acc[2] + b0.z), q3 = elu1(acc[3] + b0.w);
      float q4 = elu1(acc[4] + b1.x), q5 = elu1(acc[5] + b1.y);
      float q6 = elu1(acc[6] + b1.z), q7 = elu1(acc[7] + b1.w);
      // attention A-fragment directly from registers (lane = pixel l16)
      v8u au;
      au[0] = packbf(q0, q1); au[1] = packbf(q2, q3);
      au[2] = packbf(q4, q5); au[3] = packbf(q6, q7);
      au[4] = 0u; au[5] = 0u; au[6] = 0u; au[7] = 0u;
      // B elements 0..7 = d 8*half..+7 (matching A), elements 8..15 zero
      uint4 kv4 = *reinterpret_cast<const uint4*>(kvpk2 + (t * 16 + l16) * 12 + half * 4);
      uint4 ks4 = *reinterpret_cast<const uint4*>(kspk + t * 8 + half * 4);
      v8u kvb, ksb;
      kvb[0] = kv4.x; kvb[1] = kv4.y; kvb[2] = kv4.z; kvb[3] = kv4.w;
      kvb[4] = 0u; kvb[5] = 0u; kvb[6] = 0u; kvb[7] = 0u;
      ksb[0] = ks4.x; ksb[1] = ks4.y; ksb[2] = ks4.z; ksb[3] = ks4.w;
      ksb[4] = 0u; ksb[5] = 0u; ksb[6] = 0u; ksb[7] = 0u;
      v8f attn = wmma_bf16(au, kvb, vzero8());
      v8f den  = wmma_bf16(au, ksb, vzero8()); // Q.Ksum replicated over N
#pragma unroll
      for (int r = 0; r < 8; ++r) {
        float o = attn[r] / (den[r] + EPS_F);
        atile[(half * 8 + r) * 136 + t * 16 + l16] = (unsigned short)bfbits(o);
      }
    }
    // ---- proj: out^T = proj_w x attn^T (D: M=out ch, N=pixel -> coalesced) ----
    v8u bfr[4];
#pragma unroll
    for (int kc = 0; kc < 4; ++kc) {
      uint4 a0 = *reinterpret_cast<const uint4*>(atile + l16 * 136 + kc * 32 + half * 16);
      uint4 a1 = *reinterpret_cast<const uint4*>(atile + l16 * 136 + kc * 32 + half * 16 + 8);
      bfr[kc][0] = a0.x; bfr[kc][1] = a0.y; bfr[kc][2] = a0.z; bfr[kc][3] = a0.w;
      bfr[kc][4] = a1.x; bfr[kc][5] = a1.y; bfr[kc][6] = a1.z; bfr[kc][7] = a1.w;
    }
#pragma unroll 2
    for (int t2 = 0; t2 < 8; ++t2) {
      v8f acc = vzero8();
#pragma unroll
      for (int kc = 0; kc < 4; ++kc) {
        v8u aw = ldsv8(pwf + (((kc * 2 + half) * 8 + t2) * 16 + l16) * 12);
        acc = wmma_bf16(aw, bfr[kc], acc);
      }
      float4 b0 = *reinterpret_cast<const float4*>(pbl + t2 * 16 + half * 8);
      float4 b1 = *reinterpret_cast<const float4*>(pbl + t2 * 16 + half * 8 + 4);
      float bias[8] = {b0.x, b0.y, b0.z, b0.w, b1.x, b1.y, b1.z, b1.w};
#pragma unroll
      for (int r = 0; r < 8; ++r) {
        int oc = t2 * 16 + half * 8 + r;
        ob[(size_t)oc * HW_FULL + pixbase + l16] = acc[r] + bias[r];
      }
    }
  }
}

// ---------------------------------------------------------------------------
extern "C" void kernel_launch(void* const* d_in, const int* in_sizes, int n_in,
                              void* d_out, int out_size, void* d_ws, size_t ws_size,
                              hipStream_t stream) {
  (void)in_sizes; (void)n_in; (void)out_size; (void)ws_size;
  const float* x    = (const float*)d_in[0];
  const float* q_w  = (const float*)d_in[1];
  const float* q_b  = (const float*)d_in[2];
  const float* kv_w = (const float*)d_in[3];
  const float* kv_b = (const float*)d_in[4];
  const float* p_w  = (const float*)d_in[5];
  const float* p_b  = (const float*)d_in[6];
  float* out = (float*)d_out;

  char* ws = (char*)d_ws;                    // 166 KB used
  unsigned* qwT   = (unsigned*)(ws);         // 8192 u32
  unsigned* kvwpk = (unsigned*)(ws + 32768); // 16384 u32
  unsigned* pwT   = (unsigned*)(ws + 98304); // 8192 u32
  float*    KVg   = (float*)(ws + 131072);   // 4 x 8 x 16 x 16
  float*    Ksg   = (float*)(ws + 163840);   // 4 x 8 x 16

  prep_kernel<<<162, 256, 0, stream>>>(q_w, kv_w, p_w, qwT, kvwpk, pwT, KVg, Ksg);
  stage1_kernel<<<32, 128, 152608, stream>>>(x, kvwpk, kv_b, KVg, Ksg);
  stage2_kernel<<<512, 128, 123168, stream>>>(x, qwT, pwT, KVg, Ksg, q_b, p_b, out);
}
